// PointConvSetAbstraction_2765958938838
// MI455X (gfx1250) — compile-verified, matrix operand order checked by reading, not run on hardware
//
#include <hip/hip_runtime.h>

// ---------------------------------------------------------------------------
// PointConv Set Abstraction for MI455X (gfx1250, wave32, WMMA f16 16x16x32,
// TDM tensor_load_to_lds for the final GEMM's shared A-tile).
// Pipeline: FPS -> KNN -> gather -> GEMM1(+BN stats) -> GEMM2(+BN stats)
//           -> wn MLP (3->8->8->16, BN each) -> per-site agg WMMA -> final GEMM
// ---------------------------------------------------------------------------

typedef __attribute__((ext_vector_type(16))) _Float16 v16h;
typedef __attribute__((ext_vector_type(8)))  _Float16 v8h;
typedef __attribute__((ext_vector_type(8)))  float    v8f;
typedef __attribute__((ext_vector_type(4)))  unsigned int v4u;
typedef __attribute__((ext_vector_type(8)))  unsigned int v8u;

#define DEV static __device__ __forceinline__

constexpr int B_  = 8;
constexpr int N_  = 4096;
constexpr int S_  = 1024;           // N * 0.25
constexpr int K_  = 32;             // NSAMPLE
constexpr int D_  = 64;             // input feature channels
constexpr int C1  = 64;             // mlp layer1 out
constexpr int C2  = 128;            // mlp layer2 out
constexpr int CO  = 128;            // final linear out
constexpr int NBS = B_ * S_;        // 8192 (b,s) sites
constexpr int NS  = NBS * K_;       // 262144 conv sites
constexpr float EPSV = 1e-5f;

// --------------------------------------------------------------- utilities

__global__ void zero_kernel(float* __restrict__ p, int n) {
  int i = blockIdx.x * 256 + threadIdx.x;
  if (i < n) p[i] = 0.f;
}

__global__ void f32_to_f16_kernel(const float* __restrict__ src,
                                  _Float16* __restrict__ dst, int n) {
  int i = blockIdx.x * 256 + threadIdx.x;
  if (i < n) dst[i] = (_Float16)src[i];
}

// WMMA fragment loader: row-major f16, base = row pointer already offset by
// kcol0. Lane layout (ISA 7.12.2, 16-bit A 16x32 / symmetric B 32x16):
// components 0..7  -> k = hi + {0..7}
// components 8..15 -> k = 16 + hi + {0..7},  hi = (lane>=16)?8:0
DEV v16h load_frag_rm(const _Float16* base, int hi) {
  const v8h lo = *(const v8h*)(base + hi);
  const v8h hh = *(const v8h*)(base + 16 + hi);
  v16h r;
#pragma unroll
  for (int i = 0; i < 8; ++i) { r[i] = lo[i]; r[i + 8] = hh[i]; }
  return r;
}

// ------------------------------------------------------------------- FPS

__global__ void fps_kernel(const float* __restrict__ xyz,
                           float* __restrict__ new_xyz,   // (B*S, 3)
                           float* __restrict__ out_xyz) { // (B, 3, S) = d_out
  __shared__ float dist[N_];
  __shared__ float redv[1024];
  __shared__ int   redi[1024];
  __shared__ int   s_far;
  __shared__ float s_c[3];
  const int b = blockIdx.x, t = threadIdx.x;
  const float* xb = xyz + (size_t)b * 3 * N_;
  for (int n = t; n < N_; n += 1024) dist[n] = 1e10f;
  if (t == 0) s_far = 0;
  __syncthreads();
  for (int s = 0; s < S_; ++s) {
    if (t == 0) {
      int far = s_far;
      float cx = xb[far], cy = xb[N_ + far], cz = xb[2 * N_ + far];
      s_c[0] = cx; s_c[1] = cy; s_c[2] = cz;
      new_xyz[(size_t)(b * S_ + s) * 3 + 0] = cx;
      new_xyz[(size_t)(b * S_ + s) * 3 + 1] = cy;
      new_xyz[(size_t)(b * S_ + s) * 3 + 2] = cz;
      out_xyz[(size_t)b * 3 * S_ + 0 * S_ + s] = cx;
      out_xyz[(size_t)b * 3 * S_ + 1 * S_ + s] = cy;
      out_xyz[(size_t)b * 3 * S_ + 2 * S_ + s] = cz;
    }
    __syncthreads();
    const float cx = s_c[0], cy = s_c[1], cz = s_c[2];
    float bv = -1.f; int bi = 0;
    for (int n = t; n < N_; n += 1024) {
      float dx = xb[n] - cx, dy = xb[N_ + n] - cy, dz = xb[2 * N_ + n] - cz;
      float dn = fminf(dist[n], dx * dx + dy * dy + dz * dz);
      dist[n] = dn;
      if (dn > bv) { bv = dn; bi = n; }   // strided loop keeps lowest n on tie
    }
    redv[t] = bv; redi[t] = bi;
    __syncthreads();
    for (int off = 512; off > 0; off >>= 1) {
      if (t < off) {
        float ov = redv[t + off]; int oi = redi[t + off];
        if (ov > redv[t] || (ov == redv[t] && oi < redi[t])) {
          redv[t] = ov; redi[t] = oi;
        }
      }
      __syncthreads();
    }
    if (t == 0) s_far = redi[0];
    __syncthreads();
  }
}

// ------------------------------------------------------------------- KNN

__global__ void knn_kernel(const float* __restrict__ xyz,
                           const float* __restrict__ new_xyz,
                           int* __restrict__ nbr,        // (B*S, K)
                           float* __restrict__ offs) {   // (NS, 3)
  __shared__ float d2[N_];
  __shared__ float redv[256];
  __shared__ int   redi[256];
  const int bs = blockIdx.x, t = threadIdx.x;
  const int b = bs >> 10;
  const float* xb = xyz + (size_t)b * 3 * N_;
  const float cx = new_xyz[(size_t)bs * 3 + 0];
  const float cy = new_xyz[(size_t)bs * 3 + 1];
  const float cz = new_xyz[(size_t)bs * 3 + 2];
  for (int n = t; n < N_; n += 256) {
    float dx = xb[n] - cx, dy = xb[N_ + n] - cy, dz = xb[2 * N_ + n] - cz;
    d2[n] = dx * dx + dy * dy + dz * dz;
  }
  __syncthreads();
  for (int k = 0; k < K_; ++k) {
    float bv = 3.4e38f; int bi = 0;
    for (int n = t; n < N_; n += 256) {
      float v = d2[n];
      if (v < bv) { bv = v; bi = n; }
    }
    redv[t] = bv; redi[t] = bi;
    __syncthreads();
    for (int off = 128; off > 0; off >>= 1) {
      if (t < off) {
        float ov = redv[t + off]; int oi = redi[t + off];
        if (ov < redv[t] || (ov == redv[t] && oi < redi[t])) {
          redv[t] = ov; redi[t] = oi;
        }
      }
      __syncthreads();
    }
    if (t == 0) {
      int n = redi[0];
      size_t site = (size_t)bs * K_ + k;
      nbr[site] = n;
      offs[site * 3 + 0] = xb[n] - cx;
      offs[site * 3 + 1] = xb[N_ + n] - cy;
      offs[site * 3 + 2] = xb[2 * N_ + n] - cz;
      d2[n] = 3.4e38f;
    }
    __syncthreads();
  }
}

// ----------------------------------------------------------- feature gather

__global__ void gather_feat_kernel(const float* __restrict__ points,
                                   const int* __restrict__ nbr,
                                   _Float16* __restrict__ Xh) { // (NS, 64)
  size_t idx = (size_t)blockIdx.x * 256 + threadIdx.x;  // over NS*64
  int site = (int)(idx >> 6), c = (int)(idx & 63);
  int b = site >> 15;                                   // site / (S*K)
  int n = nbr[site];
  Xh[idx] = (_Float16)points[(size_t)b * D_ * N_ + (size_t)c * N_ + n];
}

// --------------------------------------------------- GEMM1: X@W1^T (+stats)

__global__ void gemm1_kernel(const _Float16* __restrict__ A,   // (NS,64)
                             const _Float16* __restrict__ W,   // (64,64)
                             const float* __restrict__ bias,
                             _Float16* __restrict__ Z,         // (NS,64) raw
                             float* __restrict__ g_sum, float* __restrict__ g_sq) {
  __shared__ float s_sum[C1], s_sq[C1];
  __shared__ _Float16 s_tile[32 * 64];   // block output tile (2 m-tiles x 64)
  const int t = threadIdx.x;
  if (t < C1) { s_sum[t] = 0.f; s_sq[t] = 0.f; }
  __syncthreads();
  const int wv = t >> 5, lane = t & 31;
  const int mt = blockIdx.x * 2 + (wv >> 2);
  const int nt = wv & 3;
  const int m0 = mt * 16, n0 = nt * 16;
  const int lrow = lane & 15, hi = (lane >> 4) * 8;
  const int col = n0 + lrow;
  v8f acc = {};
#pragma unroll
  for (int kb = 0; kb < 2; ++kb) {
    v16h a = load_frag_rm(A + (size_t)(m0 + lrow) * 64 + kb * 32, hi);
    v16h bf = load_frag_rm(W + (size_t)col * 64 + kb * 32, hi);
    acc = __builtin_amdgcn_wmma_f32_16x16x32_f16(false, a, false, bf,
                                                 (short)0, acc, false, false);
  }
  const float bb = bias[col];
  const int rbase = (wv >> 2) * 16;
  float ps = 0.f, pq = 0.f;
#pragma unroll
  for (int r = 0; r < 8; ++r) {
    float v = acc[r] + bb;
    ps += v; pq += v * v;
    s_tile[(rbase + r + hi) * 64 + col] = (_Float16)v;
  }
  atomicAdd(&s_sum[col], ps);
  atomicAdd(&s_sq[col], pq);
  __syncthreads();
  if (t < C1) { atomicAdd(&g_sum[t], s_sum[t]); atomicAdd(&g_sq[t], s_sq[t]); }
  // coalesced b128 store of the 32x64 tile
  {
    int h = t * 8;
    int row = h >> 6, c0 = h & 63;
    *(v8h*)&Z[(size_t)(blockIdx.x * 32 + row) * 64 + c0] = *(const v8h*)&s_tile[h];
  }
}

// ----------------------------------------------------------- BN finalize

__global__ void bn_finalize_kernel(const float* __restrict__ sum,
                                   const float* __restrict__ sq,
                                   const float* __restrict__ g,
                                   const float* __restrict__ bt,
                                   float* __restrict__ scale,
                                   float* __restrict__ shift, int C) {
  int c = threadIdx.x;
  if (c < C) {
    const float inv = 1.0f / (float)NS;
    float m = sum[c] * inv;
    float v = sq[c] * inv - m * m;
    float s = g[c] * rsqrtf(v + EPSV);
    scale[c] = s;
    shift[c] = bt[c] - m * s;
  }
}

// ------------------------------- GEMM2: bn_relu(Z1)@W2^T (+stats), fused BN

DEV v16h load_frag_bnrelu(const _Float16* base, int hi, int kcol0,
                          const float* __restrict__ scale,
                          const float* __restrict__ shift) {
  const v8h lo = *(const v8h*)(base + hi);
  const v8h hh = *(const v8h*)(base + 16 + hi);
  v16h r;
#pragma unroll
  for (int i = 0; i < 8; ++i) {
    int k0 = kcol0 + hi + i;
    int k1 = kcol0 + 16 + hi + i;
    r[i]     = (_Float16)fmaxf(0.f, (float)lo[i] * scale[k0] + shift[k0]);
    r[i + 8] = (_Float16)fmaxf(0.f, (float)hh[i] * scale[k1] + shift[k1]);
  }
  return r;
}

__global__ void gemm2_kernel(const _Float16* __restrict__ Z1,  // (NS,64) raw
                             const _Float16* __restrict__ W,   // (128,64)
                             const float* __restrict__ bias,
                             const float* __restrict__ scale1,
                             const float* __restrict__ shift1,
                             _Float16* __restrict__ Z2,        // (NS,128) raw
                             float* __restrict__ g_sum, float* __restrict__ g_sq) {
  __shared__ float s_sum[C2], s_sq[C2];
  __shared__ float s_sc[C1], s_sh[C1];
  __shared__ _Float16 s_tile[16 * C2];   // block output tile (16 x 128)
  const int t = threadIdx.x;
  if (t < C2) { s_sum[t] = 0.f; s_sq[t] = 0.f; }
  if (t < C1) { s_sc[t] = scale1[t]; s_sh[t] = shift1[t]; }
  __syncthreads();
  const int wv = t >> 5, lane = t & 31;
  const int mt = blockIdx.x, nt = wv;                  // 8 n-tiles per block
  const int m0 = mt * 16, n0 = nt * 16;
  const int lrow = lane & 15, hi = (lane >> 4) * 8;
  const int col = n0 + lrow;
  v8f acc = {};
#pragma unroll
  for (int kb = 0; kb < 2; ++kb) {
    v16h a = load_frag_bnrelu(Z1 + (size_t)(m0 + lrow) * 64 + kb * 32, hi,
                              kb * 32, s_sc, s_sh);
    v16h bf = load_frag_rm(W + (size_t)col * 64 + kb * 32, hi);
    acc = __builtin_amdgcn_wmma_f32_16x16x32_f16(false, a, false, bf,
                                                 (short)0, acc, false, false);
  }
  const float bb = bias[col];
  float ps = 0.f, pq = 0.f;
#pragma unroll
  for (int r = 0; r < 8; ++r) {
    float v = acc[r] + bb;
    ps += v; pq += v * v;
    s_tile[(r + hi) * C2 + col] = (_Float16)v;
  }
  atomicAdd(&s_sum[col], ps);
  atomicAdd(&s_sq[col], pq);
  __syncthreads();
  if (t < C2) { atomicAdd(&g_sum[t], s_sum[t]); atomicAdd(&g_sq[t], s_sq[t]); }
  // coalesced b128 store of the 16x128 tile
  {
    int h = t * 8;
    int row = h >> 7, c0 = h & 127;
    *(v8h*)&Z2[(size_t)(m0 + row) * C2 + c0] = *(const v8h*)&s_tile[h];
  }
}

// ------------------------------------------------- weight-net (tiny MLPs)

__global__ void wn1_kernel(const float* __restrict__ offs,     // (NS,3)
                           const float* __restrict__ W,        // (8,3)
                           const float* __restrict__ bias,
                           float* __restrict__ hout,           // (NS,8) raw
                           float* __restrict__ g_sum, float* __restrict__ g_sq) {
  __shared__ float s_sum[8], s_sq[8];
  int t = threadIdx.x;
  if (t < 8) { s_sum[t] = 0.f; s_sq[t] = 0.f; }
  __syncthreads();
  size_t site = (size_t)blockIdx.x * 256 + t;
  float x0 = offs[site * 3 + 0], x1 = offs[site * 3 + 1], x2 = offs[site * 3 + 2];
#pragma unroll
  for (int o = 0; o < 8; ++o) {
    float v = W[o * 3] * x0 + W[o * 3 + 1] * x1 + W[o * 3 + 2] * x2 + bias[o];
    hout[site * 8 + o] = v;
    atomicAdd(&s_sum[o], v);
    atomicAdd(&s_sq[o], v * v);
  }
  __syncthreads();
  if (t < 8) { atomicAdd(&g_sum[t], s_sum[t]); atomicAdd(&g_sq[t], s_sq[t]); }
}

template <int CIN, int COUT>
__global__ void wn_layer_kernel(const float* __restrict__ hin, // (NS,CIN) raw
                                const float* __restrict__ W,   // (COUT,CIN)
                                const float* __restrict__ bias,
                                const float* __restrict__ scale,
                                const float* __restrict__ shift,
                                float* __restrict__ hout,      // (NS,COUT) raw
                                float* __restrict__ g_sum,
                                float* __restrict__ g_sq) {
  __shared__ float s_sum[COUT], s_sq[COUT];
  int t = threadIdx.x;
  if (t < COUT) { s_sum[t] = 0.f; s_sq[t] = 0.f; }
  __syncthreads();
  size_t site = (size_t)blockIdx.x * 256 + t;
  float a[CIN];
#pragma unroll
  for (int c = 0; c < CIN; ++c)
    a[c] = fmaxf(0.f, hin[site * CIN + c] * scale[c] + shift[c]);
#pragma unroll
  for (int o = 0; o < COUT; ++o) {
    float v = bias[o];
#pragma unroll
    for (int c = 0; c < CIN; ++c) v += W[o * CIN + c] * a[c];
    hout[site * COUT + o] = v;
    atomicAdd(&s_sum[o], v);
    atomicAdd(&s_sq[o], v * v);
  }
  __syncthreads();
  if (t < COUT) { atomicAdd(&g_sum[t], s_sum[t]); atomicAdd(&g_sq[t], s_sq[t]); }
}

// ------------------- aggregation: per (b,s)  X2^T(128x32) @ Wn(32x16) -------

__global__ void agg_kernel(const _Float16* __restrict__ Z2,   // (NS,128) raw
                           const float* __restrict__ h3,      // (NS,16) raw
                           const float* __restrict__ scale2,
                           const float* __restrict__ shift2,
                           const float* __restrict__ scalew,
                           const float* __restrict__ shiftw,
                           _Float16* __restrict__ Agg) {      // (NBS, 2048)
  __shared__ _Float16 x2T[C2 * K_];   // [d][k]
  __shared__ _Float16 wnT[16 * K_];   // [w][k]
  __shared__ _Float16 s_out[2048];    // [d][w]
  const int t = threadIdx.x, bs = blockIdx.x;
  const size_t base = (size_t)bs * K_;
  for (int idx = t; idx < C2 * K_; idx += 256) {
    int d = idx & 127, k = idx >> 7;
    float z = (float)Z2[(base + k) * C2 + d];
    x2T[d * K_ + k] = (_Float16)fmaxf(0.f, z * scale2[d] + shift2[d]);
  }
  for (int idx = t; idx < 16 * K_; idx += 256) {
    int w = idx & 15, k = idx >> 4;
    float z = h3[(base + k) * 16 + w];
    wnT[w * K_ + k] = (_Float16)fmaxf(0.f, z * scalew[w] + shiftw[w]);
  }
  __syncthreads();
  const int wv = t >> 5, lane = t & 31;
  const int lrow = lane & 15, hi = (lane >> 4) * 8;
  v16h a, bf;
  {
    const _Float16* ar = &x2T[(wv * 16 + lrow) * K_];
    const v8h lo = *(const v8h*)(ar + hi);
    const v8h hh = *(const v8h*)(ar + 16 + hi);
#pragma unroll
    for (int i = 0; i < 8; ++i) { a[i] = lo[i]; a[i + 8] = hh[i]; }
  }
  {
    const _Float16* br = &wnT[lrow * K_];
    const v8h lo = *(const v8h*)(br + hi);
    const v8h hh = *(const v8h*)(br + 16 + hi);
#pragma unroll
    for (int i = 0; i < 8; ++i) { bf[i] = lo[i]; bf[i + 8] = hh[i]; }
  }
  v8f acc = {};
  acc = __builtin_amdgcn_wmma_f32_16x16x32_f16(false, a, false, bf,
                                               (short)0, acc, false, false);
#pragma unroll
  for (int r = 0; r < 8; ++r) {
    int d = wv * 16 + r + hi;
    s_out[d * 16 + lrow] = (_Float16)acc[r];
  }
  __syncthreads();
  *(v8h*)&Agg[(size_t)bs * 2048 + t * 8] = *(const v8h*)&s_out[t * 8];
}

// ----------------- final linear: Agg(8192x2048) @ linW^T(2048x128) ---------
// A-tile (16 x 2048 f16 = 64KB) is shared by all 8 waves -> stage it in LDS
// once via the Tensor Data Mover (TDM), then read fragments with ds_load_b128.

__global__ void final_gemm_kernel(const _Float16* __restrict__ Agg,
                                  const _Float16* __restrict__ Wl, // (128,2048)
                                  const float* __restrict__ bias,
                                  float* __restrict__ out) {       // (B,128,S)
  __shared__ _Float16 sA[16 * 2048];  // 64KB A-tile
  __shared__ float s_of[16 * CO];     // output tile, [col][row] for coalescing
  const int t = threadIdx.x;
  const int wv = t >> 5, lane = t & 31;
  const int mt = blockIdx.x, nt = wv;
  const int m0 = mt * 16, n0 = nt * 16;
  const int lrow = lane & 15, hi = (lane >> 4) * 8;
  const int col = n0 + lrow;

  if (wv == 0) {
    // Tensor DMA descriptor (ISA 8.3/8.4): 2D tile, 2048 x 16, f16.
    unsigned long long ga = (unsigned long long)(uintptr_t)(Agg + (size_t)m0 * 2048);
    unsigned lds = (unsigned)(uintptr_t)&sA[0];
    v4u g0;
    g0[0] = 1u;                                   // count=1, user mode
    g0[1] = lds;                                  // lds_addr
    g0[2] = (unsigned)ga;                         // global_addr[31:0]
    g0[3] = (unsigned)(ga >> 32) | (2u << 30);    // global_addr[56:32] | type=2
    v8u g1;
    g1[0] = 1u << 16;                             // data_size=1 (2 bytes)
    g1[1] = (2048u & 0xffffu) << 16;              // tensor_dim0[15:0]
    g1[2] = (2048u >> 16) | (16u << 16);          // tensor_dim0[31:16] | tensor_dim1[15:0]
    g1[3] = (2048u << 16);                        // tensor_dim1[31:16] | tile_dim0
    g1[4] = 16u;                                  // tile_dim1=16, tile_dim2=0
    g1[5] = 2048u;                                // tensor_dim0_stride[31:0]
    g1[6] = 0u;                                   // stride0[47:32] | stride1[15:0]
    g1[7] = 0u;                                   // stride1[47:16]
    asm volatile("tensor_load_to_lds %0, %1" :: "s"(g0), "s"(g1) : "memory");
#if __has_builtin(__builtin_amdgcn_s_wait_tensorcnt)
    __builtin_amdgcn_s_wait_tensorcnt(0);
#else
    asm volatile("s_wait_tensorcnt 0x0");
#endif
  }
  __syncthreads();

  const _Float16* arow = &sA[(size_t)lrow * 2048];
  const _Float16* brow = Wl + (size_t)col * 2048;
  v8f acc = {};
  for (int kb = 0; kb < 64; ++kb) {
    __builtin_prefetch(brow + (kb + 8) * 32, 0, 0);  // global_prefetch_b8
    v16h a = load_frag_rm(arow + kb * 32, hi);       // LDS (ds_load_b128)
    v16h bf = load_frag_rm(brow + kb * 32, hi);      // global b128
    acc = __builtin_amdgcn_wmma_f32_16x16x32_f16(false, a, false, bf,
                                                 (short)0, acc, false, false);
  }
  const float bb = bias[col];
#pragma unroll
  for (int r = 0; r < 8; ++r)
    s_of[col * 16 + (r + hi)] = acc[r] + bb;
  __syncthreads();
  // coalesced f32 stores: 16 consecutive s per output channel
  {
    int f = t * 8;
    int c0 = f >> 4, so = f & 15;
    int bI = m0 >> 10, s0 = m0 & 1023;
    float* dst = out + (size_t)bI * CO * S_ + (size_t)c0 * S_ + s0 + so;
    *(float4*)dst = *(const float4*)&s_of[f];
    *(float4*)(dst + 4) = *(const float4*)&s_of[f + 4];
  }
}

// ------------------------------------------------------------------- host

extern "C" void kernel_launch(void* const* d_in, const int* in_sizes, int n_in,
                              void* d_out, int out_size, void* d_ws, size_t ws_size,
                              hipStream_t stream) {
  (void)in_sizes; (void)n_in; (void)out_size;
  const float* xyz    = (const float*)d_in[0];
  const float* points = (const float*)d_in[1];
  const float* W1  = (const float*)d_in[2];
  const float* b1  = (const float*)d_in[3];
  const float* g1  = (const float*)d_in[4];
  const float* bt1 = (const float*)d_in[5];
  const float* W2  = (const float*)d_in[6];
  const float* b2  = (const float*)d_in[7];
  const float* g2  = (const float*)d_in[8];
  const float* bt2 = (const float*)d_in[9];
  const float* w0W = (const float*)d_in[10];
  const float* w0b = (const float*)d_in[11];
  const float* w0g = (const float*)d_in[12];
  const float* w0t = (const float*)d_in[13];
  const float* w1W = (const float*)d_in[14];
  const float* w1b = (const float*)d_in[15];
  const float* w1g = (const float*)d_in[16];
  const float* w1t = (const float*)d_in[17];
  const float* w2W = (const float*)d_in[18];
  const float* w2b = (const float*)d_in[19];
  const float* w2g = (const float*)d_in[20];
  const float* w2t = (const float*)d_in[21];
  const float* linW = (const float*)d_in[22];
  const float* linb = (const float*)d_in[23];

  char* ws = (char*)d_ws;
  size_t off = 0;
  auto alloc = [&](size_t bytes) -> char* {
    char* p = ws + off;
    off += (bytes + 255) & ~(size_t)255;
    return p;
  };
  float*    new_xyz = (float*)alloc((size_t)NBS * 3 * 4);
  int*      nbr     = (int*)alloc((size_t)NS * 4);
  float*    offs    = (float*)alloc((size_t)NS * 3 * 4);
  _Float16* Xh      = (_Float16*)alloc((size_t)NS * 64 * 2);
  _Float16* Z1h     = (_Float16*)alloc((size_t)NS * 64 * 2);
  _Float16* Z2h     = (_Float16*)alloc((size_t)NS * 128 * 2);
  float*    h1      = (float*)alloc((size_t)NS * 8 * 4);
  float*    h2      = (float*)alloc((size_t)NS * 8 * 4);
  float*    h3      = (float*)alloc((size_t)NS * 16 * 4);
  _Float16* Aggh    = (_Float16*)alloc((size_t)NBS * 2048 * 2);
  _Float16* W1h     = (_Float16*)alloc(64 * 64 * 2);
  _Float16* W2h     = (_Float16*)alloc(128 * 64 * 2);
  _Float16* Wlh     = (_Float16*)alloc((size_t)128 * 2048 * 2);
  float*    statf   = (float*)alloc(896 * 4);
  if (off > ws_size) return;  // workspace too small: bail deterministically

  // stats buffer layout (float offsets)
  float* sum1 = statf + 0;   float* sq1 = statf + 64;
  float* sum2 = statf + 128; float* sq2 = statf + 256;
  float* sw0  = statf + 384; float* qw0 = statf + 392;
  float* sw1  = statf + 400; float* qw1 = statf + 408;
  float* sw2  = statf + 416; float* qw2 = statf + 432;
  float* sc1  = statf + 448; float* sh1 = statf + 512;
  float* sc2  = statf + 576; float* sh2 = statf + 704;
  float* scw0 = statf + 832; float* shw0 = statf + 840;
  float* scw1 = statf + 848; float* shw1 = statf + 856;
  float* scw2 = statf + 864; float* shw2 = statf + 880;

  float* out_xyz  = (float*)d_out;                        // (B,3,S)
  float* out_feat = (float*)d_out + (size_t)B_ * 3 * S_;  // (B,128,S)

  // 0) zero BN accumulators (harness does not re-poison between replays)
  zero_kernel<<<2, 256, 0, stream>>>(statf, 448);

  // 1) weights -> f16
  f32_to_f16_kernel<<<(64 * 64 + 255) / 256, 256, 0, stream>>>(W1, W1h, 64 * 64);
  f32_to_f16_kernel<<<(128 * 64 + 255) / 256, 256, 0, stream>>>(W2, W2h, 128 * 64);
  f32_to_f16_kernel<<<(128 * 2048 + 255) / 256, 256, 0, stream>>>(linW, Wlh, 128 * 2048);

  // 2) farthest point sampling (one WG per batch; serial argmax chain)
  fps_kernel<<<B_, 1024, 0, stream>>>(xyz, new_xyz, out_xyz);

  // 3) KNN + neighbor offsets (one WG per centroid)
  knn_kernel<<<NBS, 256, 0, stream>>>(xyz, new_xyz, nbr, offs);

  // 4) gather neighbor features to f16, row-major (NS,64)
  gather_feat_kernel<<<(NS * 64) / 256, 256, 0, stream>>>(points, nbr, Xh);

  // 5) mlp layer 1 GEMM + BN stats, finalize
  gemm1_kernel<<<(NS / 16) / 2, 256, 0, stream>>>(Xh, W1h, b1, Z1h, sum1, sq1);
  bn_finalize_kernel<<<1, 128, 0, stream>>>(sum1, sq1, g1, bt1, sc1, sh1, 64);

  // 6) mlp layer 2 GEMM (bn_relu fused into A-fragment load) + stats
  gemm2_kernel<<<NS / 16, 256, 0, stream>>>(Z1h, W2h, b2, sc1, sh1, Z2h, sum2, sq2);
  bn_finalize_kernel<<<1, 128, 0, stream>>>(sum2, sq2, g2, bt2, sc2, sh2, 128);

  // 7) weight-net MLP: 3->8->8->16, BN between each
  wn1_kernel<<<NS / 256, 256, 0, stream>>>(offs, w0W, w0b, h1, sw0, qw0);
  bn_finalize_kernel<<<1, 128, 0, stream>>>(sw0, qw0, w0g, w0t, scw0, shw0, 8);
  wn_layer_kernel<8, 8><<<NS / 256, 256, 0, stream>>>(h1, w1W, w1b, scw0, shw0, h2, sw1, qw1);
  bn_finalize_kernel<<<1, 128, 0, stream>>>(sw1, qw1, w1g, w1t, scw1, shw1, 8);
  wn_layer_kernel<8, 16><<<NS / 256, 256, 0, stream>>>(h2, w2W, w2b, scw1, shw1, h3, sw2, qw2);
  bn_finalize_kernel<<<1, 128, 0, stream>>>(sw2, qw2, w2g, w2t, scw2, shw2, 16);

  // 8) per-(b,s) aggregation via WMMA (LDS-staged, bn_relu fused)
  agg_kernel<<<NBS, 256, 0, stream>>>(Z2h, h3, sc2, sh2, scw2, shw2, Aggh);

  // 9) final linear GEMM (TDM-staged A tile) -> (B,128,S) transposed output
  final_gemm_kernel<<<NBS / 16, 256, 0, stream>>>(Aggh, Wlh, linb, out_feat);
}
